// DecoderMemLM_24541443129595
// MI455X (gfx1250) — matrix operand order repaired and checked by reading, not compiled, not run on hardware
//
#include <hip/hip_runtime.h>
#include <hip/hip_bf16.h>
#include <math.h>

typedef __attribute__((ext_vector_type(16))) _Float16 v16h;
typedef __attribute__((ext_vector_type(8)))  float    v8f;

#define S_LEN 1024
#define DMODEL 1024
#define NHEAD 16
#define DHEAD 64
#define FFDIM 4096
#define BATCH 2
#define VOCAB 32000
#define NLAYER 4
#define PAD_TOK 3
#define CTXLEN 1025

// ---------------------------------------------------------------------------
// GELU (tanh approximation, matches jax.nn.gelu approximate=True)
// ---------------------------------------------------------------------------
__device__ __forceinline__ float gelu_f(float x) {
    float x3 = x * x * x;
    return 0.5f * x * (1.0f + tanhf(0.7978845608028654f * (x + 0.044715f * x3)));
}

// ---------------------------------------------------------------------------
// f32 -> f16 streaming conversion (grid-stride)
// ---------------------------------------------------------------------------
__global__ __launch_bounds__(256)
void cvt16_kernel(const float* __restrict__ in, _Float16* __restrict__ out, long n) {
    long i = (long)blockIdx.x * 256 + threadIdx.x;
    long stride = (long)gridDim.x * 256;
    for (; i < n; i += stride) out[i] = (_Float16)in[i];
}

// ---------------------------------------------------------------------------
// WMMA GEMM:  out = act(A @ B + bias) + res1 + res2
//   A: MxK f16 row-major  -> LDS via global_load_async_to_lds_b128 (ASYNCcnt)
//   B: KxN f16 row-major  -> registers, then transposed into LDS after WMMAs
//   Double-buffered LDS, one barrier per k-step; software pipelined.
//   block tile 128x128, BK=32, 256 threads = 8 waves, wave tile 32x64
// ---------------------------------------------------------------------------
__global__ __launch_bounds__(256)
void gemm_wmma_kernel(const _Float16* __restrict__ A16, const _Float16* __restrict__ B16,
                      const float* __restrict__ bias,
                      const float* __restrict__ res1, const float* __restrict__ res2,
                      float* __restrict__ C, _Float16* __restrict__ C16,
                      int M, int N, int K, int act) {
    __shared__ _Float16 As[2][128][32];   // [buf][m][k]  (row-major == memory)
    __shared__ _Float16 BsT[2][128][32];  // [buf][n][k]  (transposed)

    const int tid   = threadIdx.x;
    const int wave  = tid >> 5;
    const int lane  = tid & 31;
    const int waveM = wave >> 1;   // 0..3
    const int waveN = wave & 1;    // 0..1
    const int lrow  = lane & 15;   // 0..15
    const int lhalf = lane >> 4;   // 0..1

    const int m0 = blockIdx.y * 128;
    const int n0 = blockIdx.x * 128;

    // per-thread staging coordinates (512 chunks of 8 halfs per tile, 2/thread)
    int rA[2], cA[2], rB[2], cB[2];
    const _Float16* aPtr[2];
    const _Float16* bPtr[2];
    unsigned ldsA[2];
    const size_t bStep = (size_t)32 * N;
#pragma unroll
    for (int p = 0; p < 2; p++) {
        int chA = p * 256 + tid;
        rA[p] = chA >> 2; cA[p] = (chA & 3) * 8;
        int chB = p * 256 + tid;
        rB[p] = chB >> 4; cB[p] = (chB & 15) * 8;
        aPtr[p] = A16 + (size_t)(m0 + rA[p]) * K + cA[p];
        bPtr[p] = B16 + (size_t)rB[p] * N + n0 + cB[p];
        ldsA[p] = (unsigned)(size_t)&As[0][rA[p]][cA[p]];
    }
    const unsigned aBufStride = 128 * 32 * 2;   // bytes per A buffer

    v8f acc[2][4];
#pragma unroll
    for (int i = 0; i < 2; i++)
#pragma unroll
        for (int j = 0; j < 4; j++) acc[i][j] = (v8f){0.f,0.f,0.f,0.f,0.f,0.f,0.f,0.f};

    // ---- prologue: stage tile 0 into buffer 0 ----
#pragma unroll
    for (int p = 0; p < 2; p++) {
        asm volatile("global_load_async_to_lds_b128 %0, %1, off"
                     :: "v"(ldsA[p]), "v"(aPtr[p]) : "memory");
        aPtr[p] += 32;
    }
#pragma unroll
    for (int p = 0; p < 2; p++) {
        union { uint4 u; _Float16 h[8]; } w;
        w.u = *(const uint4*)bPtr[p];
#pragma unroll
        for (int i = 0; i < 8; i++) BsT[0][cB[p] + i][rB[p]] = w.h[i];
        bPtr[p] += bStep;
    }

    int cur = 0;
    for (int k0 = 0; k0 < K; k0 += 32) {
        const bool more = (k0 + 32 < K);
        const int nxt = cur ^ 1;

        asm volatile("s_wait_asynccnt 0x0" ::: "memory");
        __syncthreads();

        // ---- kick off tile t+1: async A copies + B global loads ----
        union { uint4 u; _Float16 h[8]; } w0, w1;
        if (more) {
#pragma unroll
            for (int p = 0; p < 2; p++) {
                unsigned la = ldsA[p] + (unsigned)nxt * aBufStride;
                asm volatile("global_load_async_to_lds_b128 %0, %1, off"
                             :: "v"(la), "v"(aPtr[p]) : "memory");
                aPtr[p] += 32;
            }
            w0.u = *(const uint4*)bPtr[0];
            w1.u = *(const uint4*)bPtr[1];
            if (k0 + 64 < K) __builtin_prefetch(bPtr[0] + bStep, 0, 1);
        }

        // ---- fragments from buffer `cur` ----
        v16h afrag[2];
#pragma unroll
        for (int fm = 0; fm < 2; fm++) {
            int row  = waveM * 32 + fm * 16 + lrow;
            int base = lhalf * 8;
#pragma unroll
            for (int e = 0; e < 8; e++) {
                afrag[fm][e]     = As[cur][row][base + e];
                afrag[fm][8 + e] = As[cur][row][16 + base + e];
            }
        }
        v16h bfrag[4];
#pragma unroll
        for (int fn = 0; fn < 4; fn++) {
            int col  = waveN * 64 + fn * 16 + lrow;
            int base = lhalf * 8;
#pragma unroll
            for (int e = 0; e < 16; e++) bfrag[fn][e] = BsT[cur][col][(e < 8) ? (base + e) : (16 + base + e - 8)];
        }
        // ---- 8 WMMAs ----
#pragma unroll
        for (int fm = 0; fm < 2; fm++)
#pragma unroll
            for (int fn = 0; fn < 4; fn++)
                acc[fm][fn] = __builtin_amdgcn_wmma_f32_16x16x32_f16(
                    false, afrag[fm], false, bfrag[fn],
                    (short)0, acc[fm][fn], false, false);

        // ---- land B tile t+1 into LDS (loadcnt wait happens here, after WMMAs) ----
        if (more) {
#pragma unroll
            for (int i = 0; i < 8; i++) BsT[nxt][cB[0] + i][rB[0]] = w0.h[i];
#pragma unroll
            for (int i = 0; i < 8; i++) BsT[nxt][cB[1] + i][rB[1]] = w1.h[i];
            bPtr[0] += bStep;
            bPtr[1] += bStep;
        }
        cur = nxt;
    }

    // ---- epilogue ----
#pragma unroll
    for (int fm = 0; fm < 2; fm++) {
#pragma unroll
        for (int fn = 0; fn < 4; fn++) {
            int mb = m0 + waveM * 32 + fm * 16;
            int nb = n0 + waveN * 64 + fn * 16 + lrow;
            float bv = bias ? bias[nb] : 0.0f;
#pragma unroll
            for (int r = 0; r < 8; r++) {
                int m = mb + r + 8 * lhalf;
                size_t idx = (size_t)m * N + nb;
                float v = acc[fm][fn][r] + bv;
                if (act) v = gelu_f(v);
                if (res1) v += res1[idx];
                if (res2) v += res2[idx];
                if (C)   C[idx]   = v;
                if (C16) C16[idx] = (_Float16)v;
            }
        }
    }
}

// ---------------------------------------------------------------------------
// LayerNorm over D=1024, one block (256 threads) per row; f32 + f16 outputs
// ---------------------------------------------------------------------------
__global__ __launch_bounds__(256)
void ln_kernel(const float* __restrict__ x, const float* __restrict__ g,
               const float* __restrict__ b, float* __restrict__ out,
               _Float16* __restrict__ out16) {
    __shared__ float red[256];
    const int row = blockIdx.x;
    const int tid = threadIdx.x;
    const float* xr = x + (size_t)row * DMODEL;

    float vloc[4];
    float s = 0.f;
#pragma unroll
    for (int i = 0; i < 4; i++) { vloc[i] = xr[tid + i * 256]; s += vloc[i]; }
    red[tid] = s; __syncthreads();
    for (int o = 128; o > 0; o >>= 1) { if (tid < o) red[tid] += red[tid + o]; __syncthreads(); }
    float mean = red[0] * (1.0f / DMODEL);
    __syncthreads();

    float vs = 0.f;
#pragma unroll
    for (int i = 0; i < 4; i++) { float d = vloc[i] - mean; vs += d * d; }
    red[tid] = vs; __syncthreads();
    for (int o = 128; o > 0; o >>= 1) { if (tid < o) red[tid] += red[tid + o]; __syncthreads(); }
    float inv = rsqrtf(red[0] * (1.0f / DMODEL) + 1e-5f);

#pragma unroll
    for (int i = 0; i < 4; i++) {
        int c = tid + i * 256;
        float o = (vloc[i] - mean) * inv * g[c] + b[c];
        out[(size_t)row * DMODEL + c]   = o;
        out16[(size_t)row * DMODEL + c] = (_Float16)o;
    }
}

// ---------------------------------------------------------------------------
// Embedding gather: x[b,s,:] = emb[text[b,s],:]
// ---------------------------------------------------------------------------
__global__ __launch_bounds__(256)
void embed_kernel(const int* __restrict__ text, const float* __restrict__ emb,
                  float* __restrict__ x) {
    const int row = blockIdx.x;
    const int b = row >> 10, s = row & 1023;
    const int tok = text[b * CTXLEN + s];
    const float* e = emb + (size_t)tok * DMODEL;
    float* o = x + (size_t)row * DMODEL;
    for (int i = threadIdx.x; i < DMODEL; i += 256) o[i] = e[i];
}

// ---------------------------------------------------------------------------
// Relative positional encodings (f16, GEMM A-operand only)
// ---------------------------------------------------------------------------
__global__ __launch_bounds__(256)
void rel_kernel(_Float16* __restrict__ rel16) {
    const int s = blockIdx.x;
    const float pos = (float)(S_LEN - 1 - s);
    for (int i = threadIdx.x; i < DMODEL; i += 256) {
        int f = (i < 512) ? i : i - 512;
        float invf = expf(-(2.0f * (float)f / (float)DMODEL) * 9.210340371976184f); // ln(1e4)
        float a = pos * invf;
        rel16[(size_t)s * DMODEL + i] = (_Float16)((i < 512) ? sinf(a) : cosf(a));
    }
}

// ---------------------------------------------------------------------------
// Attention: flash-style, one wave per query row, 8 rows per block.
//   scores[q,k] = ((q+u).k + (q+v).r[S-1-q+k]) * scale, causal + key-pad mask
//   online softmax; full-masked rows -> 0. Output f16 (GEMM A-operand).
// grid: (S/8, H, B), 256 threads
// ---------------------------------------------------------------------------
__global__ __launch_bounds__(256)
void attn_kernel(const float* __restrict__ qbuf, const float* __restrict__ kbuf,
                 const float* __restrict__ vbuf, const float* __restrict__ rbuf,
                 const float* __restrict__ u, const float* __restrict__ v,
                 const int* __restrict__ text, _Float16* __restrict__ obuf16) {
    __shared__ float qus[8][DHEAD];
    __shared__ float qvs[8][DHEAD];
    __shared__ float Ks[32][DHEAD];
    __shared__ float Vs[32][DHEAD];

    const int tid  = threadIdx.x;
    const int wave = tid >> 5;
    const int lane = tid & 31;
    const int q0 = blockIdx.x * 8;
    const int h  = blockIdx.y;
    const int b  = blockIdx.z;
    const int q  = q0 + wave;

    for (int i = tid; i < 8 * DHEAD; i += 256) {
        int r = i >> 6, d = i & 63;
        float qv_ = qbuf[(size_t)(b * S_LEN + q0 + r) * DMODEL + h * DHEAD + d];
        qus[r][d] = qv_ + u[h * DHEAD + d];
        qvs[r][d] = qv_ + v[h * DHEAD + d];
    }

    float m_run = -1e30f, sum = 0.f, acc0 = 0.f, acc1 = 0.f;
    const int kend = q0 + 7;

    for (int kt = 0; kt <= kend; kt += 32) {
        __syncthreads();
        for (int i = tid; i < 512; i += 256) {
            int r = i >> 4, c = (i & 15) * 4;
            size_t gidx = (size_t)(b * S_LEN + kt + r) * DMODEL + h * DHEAD + c;
            *(float4*)&Ks[r][c] = *(const float4*)&kbuf[gidx];
            *(float4*)&Vs[r][c] = *(const float4*)&vbuf[gidx];
        }
        __syncthreads();

        const int k = kt + lane;
        float s = -1e30f;
        bool valid = (k <= q) && (text[b * CTXLEN + k] != PAD_TOK);
        if (valid) {
            float sac = 0.f, sbd = 0.f;
            const float* rrow = rbuf + (size_t)(S_LEN - 1 - q + k) * DMODEL + h * DHEAD;
#pragma unroll 8
            for (int d = 0; d < DHEAD; d++) {
                sac = fmaf(qus[wave][d], Ks[lane][d], sac);
                sbd = fmaf(qvs[wave][d], rrow[d], sbd);
            }
            s = (sac + sbd) * 0.125f;
        }
        float mt = s;
#pragma unroll
        for (int o = 16; o > 0; o >>= 1) mt = fmaxf(mt, __shfl_xor(mt, o, 32));
        float m_new = fmaxf(m_run, mt);
        if (m_new > -1e29f) {
            float corr = expf(m_run - m_new);
            float p = (s > -1e29f) ? expf(s - m_new) : 0.f;
            float psum = p;
#pragma unroll
            for (int o = 16; o > 0; o >>= 1) psum += __shfl_xor(psum, o, 32);
            sum = sum * corr + psum;
            acc0 *= corr; acc1 *= corr;
#pragma unroll
            for (int j = 0; j < 32; j++) {
                float pj = __shfl(p, j, 32);
                acc0 = fmaf(pj, Vs[j][lane], acc0);
                acc1 = fmaf(pj, Vs[j][lane + 32], acc1);
            }
            m_run = m_new;
        }
    }

    float o0 = (sum > 0.f) ? acc0 / sum : 0.f;
    float o1 = (sum > 0.f) ? acc1 / sum : 0.f;
    size_t ob = (size_t)(b * S_LEN + q) * DMODEL + h * DHEAD;
    obuf16[ob + lane]      = (_Float16)o0;
    obuf16[ob + lane + 32] = (_Float16)o1;
}

// ---------------------------------------------------------------------------
// host-side launch
// ---------------------------------------------------------------------------
extern "C" void kernel_launch(void* const* d_in, const int* in_sizes, int n_in,
                              void* d_out, int out_size, void* d_ws, size_t ws_size,
                              hipStream_t stream) {
    const int*   text = (const int*)  d_in[0];
    const float* emb  = (const float*)d_in[1];
    const float* u    = (const float*)d_in[2];
    const float* v    = (const float*)d_in[3];
    const float* Wq   = (const float*)d_in[4];
    const float* Wk   = (const float*)d_in[5];
    const float* Wv   = (const float*)d_in[6];
    const float* Wr   = (const float*)d_in[7];
    const float* Wfc  = (const float*)d_in[8];
    const float* bfc  = (const float*)d_in[9];
    const float* ln1g = (const float*)d_in[10];
    const float* ln1b = (const float*)d_in[11];
    const float* ln2g = (const float*)d_in[12];
    const float* ln2b = (const float*)d_in[13];
    const float* W1   = (const float*)d_in[14];
    const float* b1   = (const float*)d_in[15];
    const float* W2   = (const float*)d_in[16];
    const float* b2   = (const float*)d_in[17];
    const float* lnfg = (const float*)d_in[18];
    const float* lnfb = (const float*)d_in[19];
    const float* Wlm  = (const float*)d_in[20];
    const float* blm  = (const float*)d_in[21];

    const int M = BATCH * S_LEN;                 // 2048
    const size_t MD = (size_t)M * DMODEL;        // 2,097,152
    const size_t SD = (size_t)S_LEN * DMODEL;    // 1,048,576

    char* wp = (char*)d_ws;
    auto carve = [&](size_t bytes) -> char* {
        char* r = wp; wp += (bytes + 255) & ~(size_t)255; return r;
    };
    float* x   = (float*)carve(MD * 4);
    float* xn  = (float*)carve(MD * 4);
    float* qb  = (float*)carve(MD * 4);
    float* kb  = (float*)carve(MD * 4);
    float* vb  = (float*)carve(MD * 4);
    float* rb  = (float*)carve(SD * 4);
    _Float16* xn16  = (_Float16*)carve(MD * 2);
    _Float16* o16   = (_Float16*)carve(MD * 2);
    _Float16* f116  = (_Float16*)carve((size_t)M * FFDIM * 2);
    _Float16* rel16 = (_Float16*)carve(SD * 2);
    _Float16* Wq16  = (_Float16*)carve((size_t)NLAYER * DMODEL * DMODEL * 2);
    _Float16* Wk16  = (_Float16*)carve((size_t)NLAYER * DMODEL * DMODEL * 2);
    _Float16* Wv16  = (_Float16*)carve((size_t)NLAYER * DMODEL * DMODEL * 2);
    _Float16* Wr16  = (_Float16*)carve((size_t)NLAYER * DMODEL * DMODEL * 2);
    _Float16* Wfc16 = (_Float16*)carve((size_t)NLAYER * DMODEL * DMODEL * 2);
    _Float16* W116  = (_Float16*)carve((size_t)NLAYER * DMODEL * FFDIM * 2);
    _Float16* W216  = (_Float16*)carve((size_t)NLAYER * FFDIM * DMODEL * 2);
    _Float16* Wlm16 = (_Float16*)carve((size_t)DMODEL * VOCAB * 2);

    dim3 blk(256);
    const long nDD = (long)NLAYER * DMODEL * DMODEL;
    const long nDF = (long)NLAYER * DMODEL * FFDIM;
    cvt16_kernel<<<4096, blk, 0, stream>>>(Wq,  Wq16,  nDD);
    cvt16_kernel<<<4096, blk, 0, stream>>>(Wk,  Wk16,  nDD);
    cvt16_kernel<<<4096, blk, 0, stream>>>(Wv,  Wv16,  nDD);
    cvt16_kernel<<<4096, blk, 0, stream>>>(Wr,  Wr16,  nDD);
    cvt16_kernel<<<4096, blk, 0, stream>>>(Wfc, Wfc16, nDD);
    cvt16_kernel<<<8192, blk, 0, stream>>>(W1,  W116,  nDF);
    cvt16_kernel<<<8192, blk, 0, stream>>>(W2,  W216,  nDF);
    cvt16_kernel<<<8192, blk, 0, stream>>>(Wlm, Wlm16, (long)DMODEL * VOCAB);

    embed_kernel<<<M, blk, 0, stream>>>(text, emb, x);
    rel_kernel<<<S_LEN, blk, 0, stream>>>(rel16);

    dim3 gDD(DMODEL / 128, M / 128);       // 2048x1024
    dim3 gRD(DMODEL / 128, S_LEN / 128);   // 1024x1024
    dim3 gFF(FFDIM / 128, M / 128);        // 2048x4096
    dim3 gAT(S_LEN / 8, NHEAD, BATCH);

    for (int l = 0; l < NLAYER; l++) {
        const size_t oDD = (size_t)l * DMODEL * DMODEL;
        const size_t oDF = (size_t)l * DMODEL * FFDIM;
        ln_kernel<<<M, blk, 0, stream>>>(x, ln1g + l * DMODEL, ln1b + l * DMODEL, xn, xn16);
        gemm_wmma_kernel<<<gDD, blk, 0, stream>>>(xn16, Wq16 + oDD, nullptr, nullptr, nullptr,
                                                  qb, nullptr, M, DMODEL, DMODEL, 0);
        gemm_wmma_kernel<<<gDD, blk, 0, stream>>>(xn16, Wk16 + oDD, nullptr, nullptr, nullptr,
                                                  kb, nullptr, M, DMODEL, DMODEL, 0);
        gemm_wmma_kernel<<<gDD, blk, 0, stream>>>(xn16, Wv16 + oDD, nullptr, nullptr, nullptr,
                                                  vb, nullptr, M, DMODEL, DMODEL, 0);
        gemm_wmma_kernel<<<gRD, blk, 0, stream>>>(rel16, Wr16 + oDD, nullptr, nullptr, nullptr,
                                                  rb, nullptr, S_LEN, DMODEL, DMODEL, 0);
        attn_kernel<<<gAT, blk, 0, stream>>>(qb, kb, vb, rb, u, v, text, o16);
        // x = x + (o @ Wfc + bfc) + xn
        gemm_wmma_kernel<<<gDD, blk, 0, stream>>>(o16, Wfc16 + oDD, bfc + l * DMODEL,
                                                  xn, x, x, nullptr, M, DMODEL, DMODEL, 0);
        ln_kernel<<<M, blk, 0, stream>>>(x, ln2g + l * DMODEL, ln2b + l * DMODEL, xn, xn16);
        // f1 = gelu(h @ W1 + b1)  (f16 only, feeds next GEMM)
        gemm_wmma_kernel<<<gFF, blk, 0, stream>>>(xn16, W116 + oDF, b1 + l * FFDIM,
                                                  nullptr, nullptr, nullptr, f116,
                                                  M, FFDIM, DMODEL, 1);
        // x = x + gelu(f1 @ W2 + b2)
        gemm_wmma_kernel<<<gDD, blk, 0, stream>>>(f116, W216 + oDF, b2 + l * DMODEL,
                                                  x, nullptr, x, nullptr,
                                                  M, DMODEL, FFDIM, 1);
    }

    ln_kernel<<<M, blk, 0, stream>>>(x, lnfg, lnfb, xn, xn16);
    dim3 gLM(VOCAB / 128, M / 128);
    gemm_wmma_kernel<<<gLM, blk, 0, stream>>>(xn16, Wlm16, blm, nullptr, nullptr,
                                              (float*)d_out, nullptr, M, VOCAB, DMODEL, 0);
}
